// SimCLRLoss_66374424592602
// MI455X (gfx1250) — matrix operand order repaired and checked by reading, not compiled, not run on hardware
//
#include <hip/hip_runtime.h>
#include <math.h>

#define Q_EIG   21
#define NN      32      // N_NODES
#define N_ROWS  2048
#define K_DIM   1024
#define N_GRAPH 1024
#define QR_ITERS 120

typedef __attribute__((ext_vector_type(16))) _Float16 v16h;
typedef __attribute__((ext_vector_type(8)))  _Float16 v8h;
typedef __attribute__((ext_vector_type(8)))  float    v8f;

// ---------------- helpers ----------------
__device__ __forceinline__ float wave_sum(float v) {
#pragma unroll
  for (int d = 16; d > 0; d >>= 1) v += __shfl_xor(v, d, 32);
  return v;
}

// streaming logsumexp state: (m, s) with value = log(s) + m
__device__ __forceinline__ void lse_push(float& m, float& s, float v) {
  if (v > m) { s = s * __expf(m - v) + 1.0f; m = v; }
  else       { s += __expf(v - m); }
}
__device__ __forceinline__ void lse_comb(float& m, float& s, float m2, float s2) {
  if (s2 <= 0.0f) return;
  if (s  <= 0.0f) { m = m2; s = s2; return; }
  if (m2 > m) { s = s * __expf(m - m2) + s2; m = m2; }
  else        { s += s2 * __expf(m2 - m); }
}

// ---------------- f32 -> f16 hi/lo split ----------------
__global__ void __launch_bounds__(256)
split_f16_kernel(const float* __restrict__ x, _Float16* __restrict__ hi,
                 _Float16* __restrict__ lo, int n) {
  int i = blockIdx.x * 256 + threadIdx.x;
  if (i >= n) return;
  float v = x[i];
  _Float16 h = (_Float16)v;
  hi[i] = h;
  lo[i] = (_Float16)(v - (float)h);
}

// ---------------- per-graph eigenvalues (one wave per graph) ----------------
__global__ void __launch_bounds__(32)
graph_eigs_kernel(const long long* __restrict__ edges, float* __restrict__ eigs) {
  __shared__ float M[NN][NN + 1];
  __shared__ float hv[NN];
  __shared__ float cs[NN], sn[NN];
  const int g = blockIdx.x;
  const int lane = threadIdx.x;

#pragma unroll
  for (int i = 0; i < NN; ++i) M[i][lane] = 0.0f;
  __syncthreads();

  // adjacency (directed, self-loops dropped, multi-edges accumulate)
  const long long* esrc = edges + (long long)g * 128;
  const long long* edst = esrc + 64;
  for (int e = lane; e < 64; e += 32) {
    int s = ((int)esrc[e]) & (NN - 1);
    int d = ((int)edst[e]) & (NN - 1);
    if (s != d) atomicAdd(&M[s][d], 1.0f);
  }
  __syncthreads();

  // degree (row sums), L = deg*I - A, column-normalize (norm over axis=1)
  float deg = 0.0f;
  for (int j = 0; j < NN; ++j) deg += M[lane][j];
  hv[lane] = deg;
  __syncthreads();
  for (int i = 0; i < NN; ++i)
    M[i][lane] = ((i == lane) ? hv[i] : 0.0f) - M[i][lane];
  __syncthreads();
  float nrm2 = 0.0f;
  for (int i = 0; i < NN; ++i) { float t = M[i][lane]; nrm2 += t * t; }
  float inv = 1.0f / fmaxf(sqrtf(nrm2), 1e-12f);
  for (int i = 0; i < NN; ++i) M[i][lane] *= inv;
  __syncthreads();

  // Householder reduction to Hessenberg form
  for (int k = 0; k < NN - 2; ++k) {
    float xi = (lane > k) ? M[lane][k] : 0.0f;
    float nx = sqrtf(wave_sum(xi * xi));
    float x1 = M[k + 1][k];
    float alpha = (x1 >= 0.0f) ? -nx : nx;
    float vi = xi - ((lane == k + 1) ? alpha : 0.0f);
    float vn2 = wave_sum(vi * vi);
    float beta = (vn2 > 1e-24f) ? 2.0f / vn2 : 0.0f;
    hv[lane] = vi;
    __syncthreads();
    // left: M -= beta * v (v^T M)   (lane owns a column)
    float w = 0.0f;
    for (int i = k + 1; i < NN; ++i) w += hv[i] * M[i][lane];
    for (int i = k + 1; i < NN; ++i) M[i][lane] -= beta * w * hv[i];
    __syncthreads();
    // right: M -= beta * (M v) v^T  (lane owns a row)
    float u = 0.0f;
    for (int j = k + 1; j < NN; ++j) u += M[lane][j] * hv[j];
    for (int j = k + 1; j < NN; ++j) M[lane][j] -= beta * u * hv[j];
    __syncthreads();
  }

  // unshifted Givens-QR iterations on the Hessenberg matrix
  for (int it = 0; it < QR_ITERS; ++it) {
    for (int k = 0; k < NN - 1; ++k) {
      float a = M[k][k], b = M[k + 1][k];
      float r = sqrtf(a * a + b * b);
      float c = (r > 1e-20f) ? (a / r) : 1.0f;
      float s = (r > 1e-20f) ? (b / r) : 0.0f;
      if (lane == 0) { cs[k] = c; sn[k] = s; }
      float t0 = M[k][lane], t1 = M[k + 1][lane];
      __syncthreads();
      M[k][lane]     =  c * t0 + s * t1;
      M[k + 1][lane] = -s * t0 + c * t1;
      __syncthreads();
    }
    for (int k = 0; k < NN - 1; ++k) {      // A := R * G0^T ... G30^T
      float c = cs[k], s = sn[k];
      float u0 = M[lane][k], u1 = M[lane][k + 1];
      M[lane][k]     =  c * u0 + s * u1;
      M[lane][k + 1] = -s * u0 + c * u1;
    }
    __syncthreads();
  }

  // diagonal ~ eigenvalue real parts; bitonic sort across the 32 lanes
  float ev = M[lane][lane];
#pragma unroll
  for (int size = 2; size <= 32; size <<= 1) {
#pragma unroll
    for (int stride = size >> 1; stride > 0; stride >>= 1) {
      float other = __shfl_xor(ev, stride, 32);
      bool dirUp = ((lane & size) == 0);
      bool isLow = ((lane & stride) == 0);
      float mn = fminf(ev, other), mx = fmaxf(ev, other);
      ev = (dirUp == isLow) ? mn : mx;
    }
  }
  if (lane < Q_EIG) eigs[g * Q_EIG + lane] = ev;
}

// ---------------- distance softmax statistics ----------------
__global__ void __launch_bounds__(256)
dist_stats_kernel(const float* __restrict__ eigs, float* __restrict__ part) {
  __shared__ float ei[Q_EIG];
  __shared__ float rm[8], rs[8];
  const int i = blockIdx.x;
  const int t = threadIdx.x;
  if (t < Q_EIG) ei[t] = eigs[(i & (N_GRAPH - 1)) * Q_EIG + t];
  __syncthreads();

  float m = -INFINITY, s = 0.0f;
  for (int j = t; j < N_ROWS; j += 256) {
    if (j == i) continue;                               // off-diagonal only
    const float* ej = eigs + (j & (N_GRAPH - 1)) * Q_EIG;
    float d2 = 0.0f;
#pragma unroll
    for (int q = 0; q < Q_EIG; ++q) { float df = ei[q] - ej[q]; d2 += df * df; }
    lse_push(m, s, sqrtf(d2));
  }
  int lane = t & 31, w = t >> 5;
#pragma unroll
  for (int d = 16; d > 0; d >>= 1) {
    float m2 = __shfl_xor(m, d, 32), s2 = __shfl_xor(s, d, 32);
    lse_comb(m, s, m2, s2);
  }
  if (lane == 0) { rm[w] = m; rs[w] = s; }
  __syncthreads();
  if (t == 0) {
    float M0 = rm[0], S0 = rs[0];
    for (int ww = 1; ww < 8; ++ww) lse_comb(M0, S0, rm[ww], rs[ww]);
    part[2 * i] = M0; part[2 * i + 1] = S0;
  }
}

__global__ void __launch_bounds__(256)
reduce_ms_kernel(const float* __restrict__ part, int n, float* __restrict__ scal, int mode) {
  __shared__ float rm[8], rs[8];
  const int t = threadIdx.x;
  float m = -INFINITY, s = 0.0f;
  for (int i = t; i < n; i += 256) lse_comb(m, s, part[2 * i], part[2 * i + 1]);
  int lane = t & 31, w = t >> 5;
#pragma unroll
  for (int d = 16; d > 0; d >>= 1) {
    float m2 = __shfl_xor(m, d, 32), s2 = __shfl_xor(s, d, 32);
    lse_comb(m, s, m2, s2);
  }
  if (lane == 0) { rm[w] = m; rs[w] = s; }
  __syncthreads();
  if (t == 0) {
    float M0 = rm[0], S0 = rs[0];
    for (int ww = 1; ww < 8; ++ww) lse_comb(M0, S0, rm[ww], rs[ww]);
    if (mode == 0) { scal[0] = M0; scal[1] = S0; }        // dmax, dsum
    else           { scal[2] = logf(S0) + M0; }           // LSE of similarity
  }
}

// ---------------- 16x16 similarity tile via WMMA (hi/lo f16 split) ----------------
__device__ __forceinline__ v8f sim_tile(const _Float16* __restrict__ xhi,
                                        const _Float16* __restrict__ xlo,
                                        int rowTile, int colTile, int lane) {
  const int mIdx = lane & 15;
  const int kbA = (lane >> 4) << 3;   // A halves 0..7 -> K kbA..kbA+7, 8..15 -> K 16+kbA..
  const int kbB = (lane >> 4) << 3;   // B halves 0..7 -> K kbB..kbB+7, 8..15 -> K 16+kbB..
  const _Float16* aHiP = xhi + (size_t)(rowTile * 16 + mIdx) * K_DIM;
  const _Float16* aLoP = xlo + (size_t)(rowTile * 16 + mIdx) * K_DIM;
  const _Float16* bHiP = xhi + (size_t)(colTile * 16 + mIdx) * K_DIM;
  const _Float16* bLoP = xlo + (size_t)(colTile * 16 + mIdx) * K_DIM;
  v8f acc = {};
  for (int k0 = 0; k0 < K_DIM; k0 += 32) {
    v8h a0 = *(const v8h*)(aHiP + k0 + kbA);
    v8h a1 = *(const v8h*)(aHiP + k0 + 16 + kbA);
    v16h aHi = __builtin_shufflevector(a0, a1, 0,1,2,3,4,5,6,7,8,9,10,11,12,13,14,15);
    v8h l0 = *(const v8h*)(aLoP + k0 + kbA);
    v8h l1 = *(const v8h*)(aLoP + k0 + 16 + kbA);
    v16h aLo = __builtin_shufflevector(l0, l1, 0,1,2,3,4,5,6,7,8,9,10,11,12,13,14,15);
    v8h b0 = *(const v8h*)(bHiP + k0 + kbB);
    v8h b1 = *(const v8h*)(bHiP + k0 + 16 + kbB);
    v16h bHi = __builtin_shufflevector(b0, b1, 0,1,2,3,4,5,6,7,8,9,10,11,12,13,14,15);
    v8h c0 = *(const v8h*)(bLoP + k0 + kbB);
    v8h c1 = *(const v8h*)(bLoP + k0 + 16 + kbB);
    v16h bLo = __builtin_shufflevector(c0, c1, 0,1,2,3,4,5,6,7,8,9,10,11,12,13,14,15);
    __builtin_prefetch(aHiP + k0 + 32, 0, 1);
    __builtin_prefetch(bHiP + k0 + 32, 0, 1);
    acc = __builtin_amdgcn_wmma_f32_16x16x32_f16(false, aHi, false, bHi, (short)0, acc, false, false);
    acc = __builtin_amdgcn_wmma_f32_16x16x32_f16(false, aHi, false, bLo, (short)0, acc, false, false);
    acc = __builtin_amdgcn_wmma_f32_16x16x32_f16(false, aLo, false, bHi, (short)0, acc, false, false);
  }
#pragma unroll
  for (int r = 0; r < 8; ++r) acc[r] *= 2.0f;   // 1 / TEMPERATURE
  return acc;
}

// ---------------- pass 1: logsumexp stats of off-diag similarity ----------------
__global__ void __launch_bounds__(256)
sim_lse_kernel(const _Float16* __restrict__ xhi, const _Float16* __restrict__ xlo,
               float* __restrict__ part) {
  __shared__ float rm[8], rs[8];
  const int lane = threadIdx.x & 31;
  const int w = threadIdx.x >> 5;
  const int rowTile = blockIdx.y;
  const int colTile = blockIdx.x * 8 + w;
  v8f acc = sim_tile(xhi, xlo, rowTile, colTile, lane);

  const int col = colTile * 16 + (lane & 15);
  const int rbase = rowTile * 16 + ((lane >> 4) << 3);
  float m = -INFINITY, s = 0.0f;
#pragma unroll
  for (int r = 0; r < 8; ++r) {
    int row = rbase + r;
    if (row != col) lse_push(m, s, acc[r]);
  }
#pragma unroll
  for (int d = 16; d > 0; d >>= 1) {
    float m2 = __shfl_xor(m, d, 32), s2 = __shfl_xor(s, d, 32);
    lse_comb(m, s, m2, s2);
  }
  if (lane == 0) { rm[w] = m; rs[w] = s; }
  __syncthreads();
  if (threadIdx.x == 0) {
    float M0 = rm[0], S0 = rs[0];
    for (int ww = 1; ww < 8; ++ww) lse_comb(M0, S0, rm[ww], rs[ww]);
    int p = blockIdx.y * gridDim.x + blockIdx.x;
    part[2 * p] = M0; part[2 * p + 1] = S0;
  }
}

// ---------------- pass 2: sum exp(d - dmax) * s over off-diagonal ----------------
__global__ void __launch_bounds__(256)
sim_final_kernel(const _Float16* __restrict__ xhi, const _Float16* __restrict__ xlo,
                 const float* __restrict__ eigs, const float* __restrict__ scal,
                 float* __restrict__ part) {
  __shared__ float eRow[16][Q_EIG + 1];
  __shared__ float eCol[128][Q_EIG + 1];
  __shared__ float rsum[8];
  const int rowTile = blockIdx.y;
  const int colBase = blockIdx.x * 8;

  for (int idx = threadIdx.x; idx < 16 * Q_EIG; idx += 256) {
    int rr = idx / Q_EIG, q = idx % Q_EIG;
    eRow[rr][q] = eigs[((rowTile * 16 + rr) & (N_GRAPH - 1)) * Q_EIG + q];
  }
  for (int idx = threadIdx.x; idx < 128 * Q_EIG; idx += 256) {
    int cc = idx / Q_EIG, q = idx % Q_EIG;
    eCol[cc][q] = eigs[((colBase * 16 + cc) & (N_GRAPH - 1)) * Q_EIG + q];
  }
  __syncthreads();

  const int lane = threadIdx.x & 31;
  const int w = threadIdx.x >> 5;
  const int colTile = colBase + w;
  v8f acc = sim_tile(xhi, xlo, rowTile, colTile, lane);

  const float dmax = scal[0];
  const int lcol = w * 16 + (lane & 15);
  const int col = colBase * 16 + lcol;
  const int lrbase = (lane >> 4) << 3;
  float sum = 0.0f;
#pragma unroll
  for (int r = 0; r < 8; ++r) {
    int lrow = lrbase + r;
    int row = rowTile * 16 + lrow;
    if (row == col) continue;
    float d2 = 0.0f;
#pragma unroll
    for (int q = 0; q < Q_EIG; ++q) {
      float df = eRow[lrow][q] - eCol[lcol][q];
      d2 += df * df;
    }
    sum += __expf(sqrtf(d2) - dmax) * acc[r];
  }
  sum = wave_sum(sum);
  if (lane == 0) rsum[w] = sum;
  __syncthreads();
  if (threadIdx.x == 0) {
    float tot = 0.0f;
    for (int ww = 0; ww < 8; ++ww) tot += rsum[ww];
    part[blockIdx.y * gridDim.x + blockIdx.x] = tot;
  }
}

__global__ void __launch_bounds__(256)
finalize_kernel(const float* __restrict__ part, int n,
                const float* __restrict__ scal, float* __restrict__ out) {
  __shared__ float rsum[8];
  float s = 0.0f;
  for (int i = threadIdx.x; i < n; i += 256) s += part[i];
  s = wave_sum(s);
  int lane = threadIdx.x & 31, w = threadIdx.x >> 5;
  if (lane == 0) rsum[w] = s;
  __syncthreads();
  if (threadIdx.x == 0) {
    float tot = 0.0f;
    for (int ww = 0; ww < 8; ++ww) tot += rsum[ww];
    float dsum = scal[1];                      // softmax denominator of distances
    float lse  = scal[2];                      // logsumexp of similarity
    out[0] = -tot / (dsum * lse);
  }
}

// ---------------- launch ----------------
extern "C" void kernel_launch(void* const* d_in, const int* in_sizes, int n_in,
                              void* d_out, int out_size, void* d_ws, size_t ws_size,
                              hipStream_t stream) {
  (void)in_sizes; (void)n_in; (void)out_size; (void)ws_size;
  const float* x = (const float*)d_in[0];
  const long long* edges = (const long long*)d_in[1];

  char* ws = (char*)d_ws;
  float*    eigs  = (float*)(ws + 0);             // 1024*21 f32      = 86016 B
  float*    scal  = (float*)(ws + 86016);         // 8 f32
  float*    dpart = (float*)(ws + 86048);         // 2048 pairs       = 16384 B
  float*    spart = (float*)(ws + 102432);        // 2048 pairs       = 16384 B
  float*    fpart = (float*)(ws + 118816);        // 2048 f32         =  8192 B
  _Float16* xhi   = (_Float16*)(ws + 127104);     // 2048*1024 f16    = 4 MiB
  _Float16* xlo   = (_Float16*)(ws + 127104 + (size_t)N_ROWS * K_DIM * 2);

  const int nElem = N_ROWS * K_DIM;
  split_f16_kernel<<<(nElem + 255) / 256, 256, 0, stream>>>(x, xhi, xlo, nElem);
  graph_eigs_kernel<<<N_GRAPH, 32, 0, stream>>>(edges, eigs);
  dist_stats_kernel<<<N_ROWS, 256, 0, stream>>>(eigs, dpart);
  reduce_ms_kernel<<<1, 256, 0, stream>>>(dpart, N_ROWS, scal, 0);

  dim3 grid(16, 128);   // 128x128 tiles of 16x16, 8 waves (tiles) per block
  sim_lse_kernel<<<grid, 256, 0, stream>>>(xhi, xlo, spart);
  reduce_ms_kernel<<<1, 256, 0, stream>>>(spart, 2048, scal, 1);
  sim_final_kernel<<<grid, 256, 0, stream>>>(xhi, xlo, eigs, scal, fpart);
  finalize_kernel<<<1, 256, 0, stream>>>(fpart, 2048, scal, (float*)d_out);
}
